// RVQEARTTSModel_62569083568280
// MI455X (gfx1250) — matrix operand order repaired and checked by reading, not compiled, not run on hardware
//
#include <hip/hip_runtime.h>

typedef __attribute__((ext_vector_type(16))) __bf16 v16bf;
typedef __attribute__((ext_vector_type(8)))  __bf16 v8bf;
typedef __attribute__((ext_vector_type(4)))  __bf16 v4bf;
typedef __attribute__((ext_vector_type(8)))  float  v8f;
typedef __attribute__((ext_vector_type(4)))  float  v4f;

#define KDEP 72
#define CSZ  1024
#define DDIM 128
#define NTOK 32768          // B*T = 16*2048
#define WAVES 8
#define TOK_PER_WAVE 16
#define TOK_PER_BLK  128
#define NTILE (CSZ / 16)    // 64 code tiles of 16 per depth
#define TILE_ELEMS (16 * DDIM)   // 2048 bf16 per (hi|lo) tile in global

// Padded LDS rows -> per-lane row stride ≡ 4 dwords (mod 64 banks): conflict-free.
#define BROW 136            // bf16/row in staged B tiles (128 data + 8 pad = 68 dwords)
#define RROW 132            // f32/row in residual buffer  (128 data + 4 pad = 132 dwords)

// ---------------------------------------------------------------------------
// Prep: split codebooks into bf16 hi/lo (bf16x3 fp32 emulation) + ||e||^2.
// ---------------------------------------------------------------------------
__global__ __launch_bounds__(256) void rvq_prep(const float* __restrict__ embs,
                                                __bf16* __restrict__ ehi,
                                                __bf16* __restrict__ elo,
                                                float*  __restrict__ enorm)
{
    const int lane = threadIdx.x & 31;
    const int wave = threadIdx.x >> 5;
    const int row  = blockIdx.x * WAVES + wave;     // 0 .. K*C-1
    if (row >= KDEP * CSZ) return;

    const size_t base = (size_t)row * DDIM + lane * 4;
    v4f v = *(const v4f*)(embs + base);
    v4bf h, l;
    float ss = 0.f;
#pragma unroll
    for (int i = 0; i < 4; ++i) {
        float  f  = v[i];
        __bf16 hh = (__bf16)f;
        h[i] = hh;
        l[i] = (__bf16)(f - (float)hh);
        ss  += f * f;
    }
    *(v4bf*)(ehi + base) = h;
    *(v4bf*)(elo + base) = l;
#pragma unroll
    for (int m = 16; m >= 1; m >>= 1) ss += __shfl_xor(ss, m, 32);
    if (lane == 0) enorm[row] = ss;
}

// Async-stage one 16-code tile (hi + lo) into padded LDS rows: thread t moves
// the 16B chunk (t&15) of row (t>>4) for both matrices via
// global_load_async_to_lds_b128 (ASYNCcnt-tracked, no VGPR data path).
__device__ __forceinline__ void stage_tile(const __bf16* __restrict__ ehi,
                                           const __bf16* __restrict__ elo,
                                           __bf16* dst_hi, __bf16* dst_lo,
                                           size_t src_off, int tid)
{
    const int row   = tid >> 4;
    const int chunk = tid & 15;
    const int doff  = row * BROW + chunk * 8;          // padded LDS layout
    const int soff  = row * DDIM + chunk * 8;          // dense global layout
    unsigned           lh = (unsigned)(unsigned long long)(dst_hi + doff);
    unsigned           ll = (unsigned)(unsigned long long)(dst_lo + doff);
    unsigned long long gh = (unsigned long long)(ehi + src_off + soff);
    unsigned long long gl = (unsigned long long)(elo + src_off + soff);
    asm volatile("global_load_async_to_lds_b128 %0, %1, off" :: "v"(lh), "v"(gh) : "memory");
    asm volatile("global_load_async_to_lds_b128 %0, %1, off" :: "v"(ll), "v"(gl) : "memory");
}

// ---------------------------------------------------------------------------
// Main: whole 72-depth RVQ chain with r resident in LDS, codebook tiles
// double-buffered in LDS via async copies one tile ahead of the WMMAs.
// ---------------------------------------------------------------------------
__global__ __launch_bounds__(256, 1) void rvq_main(const float* __restrict__ embs,
                                                   const float* __restrict__ rin,
                                                   const __bf16* __restrict__ ehi,
                                                   const __bf16* __restrict__ elo,
                                                   const float* __restrict__ enorm,
                                                   long long* __restrict__ code)
{
    __shared__ float  rbuf[TOK_PER_BLK * RROW];          // ~66 KB residuals (padded)
    __shared__ __bf16 bstage[2][2][16 * BROW];           // 2 bufs x {hi,lo} (padded)

    const int tid  = threadIdx.x;
    const int lane = tid & 31;
    const int wave = tid >> 5;
    const int tb   = blockIdx.x * TOK_PER_BLK + wave * TOK_PER_WAVE;

    // Prologue: kick off async stage of (dep 0, tile 0) before anything else.
    stage_tile(ehi, elo, &bstage[0][0][0], &bstage[0][1][0], 0, tid);

    // Load this wave's 16 residual rows into LDS (overlaps the async stage).
    {
        const int d0 = lane * 4;
#pragma unroll
        for (int t = 0; t < TOK_PER_WAVE; ++t) {
            v4f v = *(const v4f*)(rin + (size_t)(tb + t) * DDIM + d0);
            *(v4f*)&rbuf[(wave * TOK_PER_WAVE + t) * RROW + d0] = v;
        }
    }

    const int m = lane & 15;        // row (A) / column (B,C) index
    const int g = lane >> 4;        // lane-group (K-half selector)

    for (int dep = 0; dep < KDEP; ++dep) {
        // ---- A fragments: 16x32 bf16 layout (lanes 0-15: K 8g+[0..8),16+8g+[0..8)) ----
        v16bf rh[4], rl[4];
        {
            const float* rrow = &rbuf[(wave * TOK_PER_WAVE + m) * RROW];
#pragma unroll
            for (int q = 0; q < 4; ++q) {
                const int b0 = 32 * q + 8 * g;
#pragma unroll
                for (int i = 0; i < 8; ++i) {
                    float  f0 = rrow[b0 + i];
                    float  f1 = rrow[b0 + 16 + i];
                    __bf16 h0 = (__bf16)f0, h1 = (__bf16)f1;
                    rh[q][i]     = h0;
                    rh[q][8 + i] = h1;
                    rl[q][i]     = (__bf16)(f0 - (float)h0);
                    rl[q][8 + i] = (__bf16)(f1 - (float)h1);
                }
            }
        }

        float best[8];
        int   bidx[8];
#pragma unroll
        for (int j = 0; j < 8; ++j) { best[j] = 3.4e38f; bidx[j] = 0; }

        const size_t kbase = (size_t)dep * CSZ * DDIM;
        const float* enk   = enorm + (size_t)dep * CSZ;

        for (int tile = 0; tile < NTILE; ++tile) {
            const int buf = tile & 1;

            // Buffer `buf` was staged one tile ago: drain this wave's async
            // copies, then barrier so (a) everyone sees the staged data and
            // (b) everyone is done reading the buffer we are about to refill.
            asm volatile("s_wait_asynccnt 0x0" ::: "memory");
            __syncthreads();

            // Stage the next tile (or next depth's tile 0) into the other buffer.
            if (tile + 1 < NTILE) {
                stage_tile(ehi, elo, &bstage[buf ^ 1][0][0], &bstage[buf ^ 1][1][0],
                           kbase + (size_t)(tile + 1) * TILE_ELEMS, tid);
            } else if (dep + 1 < KDEP) {
                stage_tile(ehi, elo, &bstage[buf ^ 1][0][0], &bstage[buf ^ 1][1][0],
                           kbase + (size_t)CSZ * DDIM, tid);
            }

            // ---- 12 WMMAs (bf16x3 over 4 K-chunks) from LDS-resident B tiles ----
            const __bf16* ph = &bstage[buf][0][m * BROW + 8 * g];
            const __bf16* pl = &bstage[buf][1][m * BROW + 8 * g];

            v8f acc = {};
#pragma unroll
            for (int q = 0; q < 4; ++q) {
                v8bf h0 = *(const v8bf*)(ph + 32 * q);
                v8bf h1 = *(const v8bf*)(ph + 32 * q + 16);
                v8bf l0 = *(const v8bf*)(pl + 32 * q);
                v8bf l1 = *(const v8bf*)(pl + 32 * q + 16);
                v16bf bh, bl;
#pragma unroll
                for (int i = 0; i < 8; ++i) {
                    bh[i] = h0[i]; bh[8 + i] = h1[i];
                    bl[i] = l0[i]; bl[8 + i] = l1[i];
                }
                // bf16x3: rh*eh + rh*el + rl*eh  ~= fp32 dot(r, e)
                acc = __builtin_amdgcn_wmma_f32_16x16x32_bf16(false, rl[q], false, bh,
                                                              (short)0, acc, false, false);
                acc = __builtin_amdgcn_wmma_f32_16x16x32_bf16(false, rh[q], false, bl,
                                                              (short)0, acc, false, false);
                acc = __builtin_amdgcn_wmma_f32_16x16x32_bf16(false, rh[q], false, bh,
                                                              (short)0, acc, false, false);
            }

            const int   c  = tile * 16 + m;
            const float en = enk[c];
#pragma unroll
            for (int j = 0; j < 8; ++j) {
                float dist = en - 2.0f * acc[j];     // ||e||^2 - 2 r.e
                if (dist < best[j]) { best[j] = dist; bidx[j] = c; }
            }
        }

        // ---- argmin across the 16 lanes of each group (ties -> lowest index) ----
#pragma unroll
        for (int msk = 1; msk <= 8; msk <<= 1) {
#pragma unroll
            for (int j = 0; j < 8; ++j) {
                float od = __shfl_xor(best[j], msk, 32);
                int   oi = __shfl_xor(bidx[j], msk, 32);
                if (od < best[j] || (od == best[j] && oi < bidx[j])) {
                    best[j] = od; bidx[j] = oi;
                }
            }
        }

        // ---- emit code[b,t,dep] (int64) ----
#pragma unroll
        for (int j = 0; j < 8; ++j) {
            if (lane == 0)  code[(size_t)(tb + j)     * KDEP + dep] = (long long)bidx[j];
            if (lane == 16) code[(size_t)(tb + 8 + j) * KDEP + dep] = (long long)bidx[j];
        }

        // ---- fp32 residual update from the original codebook (whole wave) ----
        // Overlaps the async stage of the next depth's first tile.
#pragma unroll 4
        for (int t = 0; t < TOK_PER_WAVE; ++t) {
            int idx = __shfl(bidx[t & 7], (t < 8) ? 0 : 16, 32);
            v4f ev  = *(const v4f*)(embs + kbase + (size_t)idx * DDIM + lane * 4);
            v4f* rp = (v4f*)&rbuf[(wave * TOK_PER_WAVE + t) * RROW + lane * 4];
            v4f rv  = *rp;
            rv -= ev;
            *rp = rv;
        }
    }
}

// ---------------------------------------------------------------------------
extern "C" void kernel_launch(void* const* d_in, const int* in_sizes, int n_in,
                              void* d_out, int out_size, void* d_ws, size_t ws_size,
                              hipStream_t stream)
{
    (void)in_sizes; (void)n_in; (void)out_size; (void)ws_size;

    const float* embs = (const float*)d_in[0];   // [72,1024,128] f32
    const float* rin  = (const float*)d_in[1];   // [16,2048,128] f32
    long long*   code = (long long*)d_out;       // [16,2048,72]  i64

    // Workspace layout: e_hi bf16 | e_lo bf16 | ||e||^2 f32  (~38 MB)
    char*   ws   = (char*)d_ws;
    size_t  n_e  = (size_t)KDEP * CSZ * DDIM;
    __bf16* ehi  = (__bf16*)ws;
    __bf16* elo  = (__bf16*)(ws + n_e * sizeof(__bf16));
    float*  enrm = (float*)(ws + 2 * n_e * sizeof(__bf16));

    rvq_prep<<<dim3((KDEP * CSZ) / WAVES), 256, 0, stream>>>(embs, ehi, elo, enrm);
    rvq_main<<<dim3(NTOK / TOK_PER_BLK), 256, 0, stream>>>(embs, rin, ehi, elo, enrm, code);
}
